// SPCNNClassifier_41420664603135
// MI455X (gfx1250) — matrix operand order repaired and checked by reading, not compiled, not run on hardware
//
#include <hip/hip_runtime.h>
#include <hip/hip_bf16.h>
#include <stdint.h>

// ---------------------------------------------------------------------------
// Problem constants (from reference): B=32 T=512 E=1024 H=512 L=6 C=256
// ---------------------------------------------------------------------------
#define B_   32
#define T_   512
#define E_   1024
#define H_   512
#define L_   6
#define C_   256
#define D3C  768     // 3*C  (conv concat channels)
#define G4H  2048    // 4*H  (gate width)
#define H2   1024    // 2*H  (bi-directional concat)
#define MROW 16384   // T*B rows for the big GEMMs
#define PAD_ 8       // zero-padded time rows on each side for the conv
#define TP_  (T_ + 2 * PAD_)   // 528 padded time rows
#define EPS_ 1e-5f

typedef __attribute__((ext_vector_type(16))) _Float16 h16;
typedef __attribute__((ext_vector_type(8)))  float    f8;
typedef __attribute__((ext_vector_type(4)))  unsigned u32x4;
typedef __attribute__((ext_vector_type(8)))  int      i32x8;
typedef __attribute__((ext_vector_type(4)))  int      i32x4;

// ---------------------------------------------------------------------------
// WMMA wrapper (codegen-confirmed form on gfx1250)
// ---------------------------------------------------------------------------
__device__ __forceinline__ f8 wmma16(h16 a, h16 b, f8 c) {
  // 8 args: (neg_a, A, neg_b, B, c_mod, C, reuse_a, reuse_b)
  return __builtin_amdgcn_wmma_f32_16x16x32_f16(false, a, false, b,
                                                (short)0, c, false, false);
}

// A-fragment (16x32 f16, M=rows, K=reduction) from a row-major matrix.
// ISA layout: lane<16 -> row M=lane, elems = K {0..7,16..23};
//             lane>=16 -> row M=lane-16, elems = K {8..15,24..31}.
__device__ __forceinline__ h16 load_a_tile(const _Float16* base, int ld,
                                           int r0, int k0) {
  int lane = threadIdx.x & 31;
  int half = lane >> 4;
  int row  = lane & 15;
  const _Float16* p = base + (size_t)(r0 + row) * ld + k0 + half * 8;
  union { h16 v; uint4 q[2]; } u;
  u.q[0] = *(const uint4*)(p);
  u.q[1] = *(const uint4*)(p + 16);
  return u.v;
}

// B-fragment from pre-packed buffer: packed[((kb*NT + nt)*32 + lane)*16 + j]
__device__ __forceinline__ h16 load_b_packed(const _Float16* packed, int NT,
                                             int kb, int nt) {
  int lane = threadIdx.x & 31;
  const _Float16* p = packed + (((size_t)kb * NT + nt) * 32 + lane) * 16;
  return *(const h16*)p;
}

__device__ __forceinline__ float sigmoidf_(float x) {
  return 1.0f / (1.0f + __expf(-x));
}
__device__ __forceinline__ float tanhf_(float x) {
  return 1.0f - 2.0f / (__expf(2.0f * x) + 1.0f);
}

// ---------------------------------------------------------------------------
// Tensor Data Mover: 1-row tile load (contiguous bytes) global -> LDS.
// D# per CDNA5 ISA: group0 = {flags, lds_addr, gaddr_lo,
// gaddr_hi|type=2}; group1 carries data_size=8B, tensor/tile dim0 in 8B units.
// LDS byte address = low 32 bits of the generic pointer (aperture rule).
// ---------------------------------------------------------------------------
__device__ __forceinline__ void tdm_load_to_lds(const void* gsrc, void* ldst,
                                                unsigned bytes) {
  unsigned lds_addr = (unsigned)(uintptr_t)ldst;
  unsigned long long ga = (unsigned long long)(uintptr_t)gsrc;
  unsigned units = bytes >> 3;                     // 8-byte elements
  u32x4 g0;
  g0[0] = 1u;                                      // count=1 valid descriptor
  g0[1] = lds_addr;                                // lds_addr [63:32]
  g0[2] = (unsigned)(ga & 0xffffffffu);            // global_addr lo
  g0[3] = (unsigned)((ga >> 32) & 0x01ffffffu)     // global_addr [56:32]
          | (2u << 30);                            // type = 2 ("image")
  i32x8 g1;
  g1[0] = (int)(3u << 16);                         // data_size=3 (8B), mask=0
  g1[1] = (int)((units & 0xffffu) << 16);          // tensor_dim0[15:0] @ b48
  g1[2] = (int)(((units >> 16) & 0xffffu)          // tensor_dim0[31:16]
          | (1u << 16));                           // tensor_dim1 = 1
  g1[3] = (int)((units & 0xffffu) << 16);          // tile_dim0 @ bits[127:112]
  g1[4] = 1;                                       // tile_dim1 = 1
  g1[5] = (int)units;                              // tensor_dim0_stride lo
  g1[6] = 0;
  g1[7] = 0;
  i32x4 z4 = {0, 0, 0, 0};
#if __clang_major__ >= 23
  i32x8 z8 = {0, 0, 0, 0, 0, 0, 0, 0};
  __builtin_amdgcn_tensor_load_to_lds(g0, g1, z4, z4, z8, 0);
#else
  __builtin_amdgcn_tensor_load_to_lds(g0, g1, z4, z4, 0);
#endif
}

// ---------------------------------------------------------------------------
// 0) f32 -> f16 cast with zero time-padding: dst shape (B, TP_, E)
// ---------------------------------------------------------------------------
__global__ void cast_pad_f16(const float* __restrict__ src,
                             _Float16* __restrict__ dst) {
  long i = (long)blockIdx.x * blockDim.x + threadIdx.x;
  if (i >= (long)B_ * TP_ * E_) return;
  int  e = (int)(i % E_);
  long r = i / E_;
  int  tp = (int)(r % TP_);
  int  b  = (int)(r / TP_);
  int  t  = tp - PAD_;
  dst[i] = (t >= 0 && t < T_)
               ? (_Float16)src[((long)b * T_ + t) * E_ + e]
               : (_Float16)0.0f;
}

// ---------------------------------------------------------------------------
// 1) Pack an (N x K) weight view into WMMA B-fragment order (f16).
//    dst[((kb*NT+nt)*32+lane)*16+j] = src[soff + n*sn + k*sk]
//    with n = nt*16 + (lane&15), k = kb*32 + (lane>>4)*8 + (j&7) + (j>>3)*16.
// ---------------------------------------------------------------------------
__global__ void pack_b_f16(const float* __restrict__ src,
                           _Float16* __restrict__ dst,
                           int Ktot, int Ntot, long sk, long sn, long soff) {
  long idx = (long)blockIdx.x * blockDim.x + threadIdx.x;
  long total = (long)Ktot * Ntot;
  if (idx >= total) return;
  int  j    = (int)(idx & 15);
  long r    = idx >> 4;
  int  lane = (int)(r & 31);
  r >>= 5;
  int NT = Ntot >> 4;
  int nt = (int)(r % NT);
  int kb = (int)(r / NT);
  int n  = nt * 16 + (lane & 15);
  int k  = kb * 32 + (lane >> 4) * 8 + (j & 7) + (j >> 3) * 16;
  dst[idx] = (_Float16)src[soff + (long)n * sn + (long)k * sk];
}

// ---------------------------------------------------------------------------
// 2) Conv bank (3 convs over E, taps 5/9/9) as shifted WMMA GEMMs on the
//    zero-padded f16 input -> branch-free inner loop. Fused BN + exact GELU,
//    output f16 (t*B+b, 768). grid = (T/16, B, 48), block = 32 (one wave).
// ---------------------------------------------------------------------------
__global__ void conv_bn_gelu(const _Float16* __restrict__ xb,   // (B,TP_,E) f16
                             const _Float16* __restrict__ wc5,  // packed, 5 taps
                             const _Float16* __restrict__ wc7,  // packed, 9 taps
                             const _Float16* __restrict__ wc9,  // packed, 9 taps
                             const float* __restrict__ b5,
                             const float* __restrict__ b7,
                             const float* __restrict__ b9,
                             const float* __restrict__ bn_gamma,
                             const float* __restrict__ bn_beta,
                             const float* __restrict__ bn_mean,
                             const float* __restrict__ bn_var,
                             _Float16* __restrict__ xc) {       // (T*B, 768)
  int t0   = blockIdx.x * 16;
  int b    = blockIdx.y;
  int cog0 = blockIdx.z * 16;
  int cv   = blockIdx.z >> 4;          // 0,1,2 -> which conv (16 tiles each)
  int nt   = blockIdx.z & 15;          // tile index inside this conv
  int taps = (cv == 0) ? 5 : 9;
  int padm = (cv == 0) ? 2 : 4;
  const _Float16* wpk = (cv == 0) ? wc5 : (cv == 1 ? wc7 : wc9);
  const float*    cb  = (cv == 0) ? b5  : (cv == 1 ? b7  : b9);

  int lane = threadIdx.x & 31;
  int half = lane >> 4;
  int row  = lane & 15;

  f8 acc = {};
#pragma unroll 1
  for (int tap = 0; tap < taps; ++tap) {
    // padded time row for this lane's output row (always in range)
    int tsp = t0 + tap - padm + PAD_ + row;
    const _Float16* arow = xb + ((size_t)b * TP_ + tsp) * E_ + half * 8;
    const _Float16* wtap = wpk + (size_t)tap * E_ * C_;
#pragma unroll 4
    for (int kb = 0; kb < (E_ >> 5); ++kb) {
      union { h16 v; uint4 q[2]; } u;
      u.q[0] = *(const uint4*)(arow + kb * 32);
      u.q[1] = *(const uint4*)(arow + kb * 32 + 16);
      h16 bf = load_b_packed(wtap, C_ >> 4, kb, nt);
      acc = wmma16(u.v, bf, acc);
    }
  }

  int cog = cog0 + (lane & 15);                 // global channel 0..767
  float scale = bn_gamma[cog] * rsqrtf(bn_var[cog] + EPS_);
  float shift = bn_beta[cog] - bn_mean[cog] * scale;
  float bias  = cb[cog - cv * C_];
#pragma unroll
  for (int r = 0; r < 8; ++r) {
    int tt = t0 + r + 8 * half;                 // D: M = r + 8*(lane>=16)
    float v = acc[r] + bias;
    v = v * scale + shift;
    v = 0.5f * v * (1.0f + erff(v * 0.70710678118654752f));
    xc[((size_t)tt * B_ + b) * D3C + cog] = (_Float16)v;
  }
}

// ---------------------------------------------------------------------------
// 3) Generic GEMM: C(MxN,f32) = A(MxK,f16 row-major) * Bpacked + bias1+bias2.
//    block = 256 (8 waves), wave w owns N-tile blockIdx.y*8+w.
// ---------------------------------------------------------------------------
__global__ void __launch_bounds__(256)
gemm_bias(const _Float16* __restrict__ A, const _Float16* __restrict__ Bp,
          const float* __restrict__ b1, const float* __restrict__ b2,
          float* __restrict__ C, int M, int K, int N) {
  int wave = threadIdx.x >> 5;
  int nt   = blockIdx.y * 8 + wave;
  int mt   = blockIdx.x;
  int lane = threadIdx.x & 31;
  int half = lane >> 4;

  f8 acc = {};
  int KB = K >> 5;
#pragma unroll 4
  for (int kb = 0; kb < KB; ++kb) {
    h16 a  = load_a_tile(A, K, mt * 16, kb * 32);
    h16 bf = load_b_packed(Bp, N >> 4, kb, nt);
    acc = wmma16(a, bf, acc);
  }
  int n = nt * 16 + (lane & 15);
  float bb = b1[n] + b2[n];
#pragma unroll
  for (int r = 0; r < 8; ++r) {
    int m = mt * 16 + r + 8 * half;
    C[(size_t)m * N + n] = acc[r] + bb;
  }
}

// ---------------------------------------------------------------------------
// 4) Persistent LSTM recurrence. 4 workgroups (dispatched as a workgroup
//    cluster) x 512 threads (16 waves). Each WG owns 128 h-columns and
//    computes all 4 gates for them; full h is exchanged each step through a
//    double-buffered global staging buffer under the cluster barrier, and
//    pulled back into LDS with a Tensor Data Mover descriptor (TENSORcnt).
// ---------------------------------------------------------------------------
__global__ void __launch_bounds__(512)
lstm_scan(const float* __restrict__ xg,        // (T*B, 2048) precomputed x-gates
          const _Float16* __restrict__ whh_pk, // packed (K=512 -> N=2048)
          _Float16* __restrict__ hstage,       // 2 * B*H f16 staging
          _Float16* __restrict__ hout,         // (T*B, 2H), slice dir*H
          int reverse, int dir) {
  __shared__ __align__(16) _Float16 h_lds[B_ * H_];   // 32 KB, full h
  __shared__ __align__(16) float    c_lds[B_ * 128];  // 16 KB, owned c slice

  const int wg   = blockIdx.x;          // 0..3, owns hcols [wg*128, wg*128+128)
  const int wave = threadIdx.x >> 5;    // 0..15
  const int gm   = wave >> 3;           // batch half: rows gm*16..gm*16+15
  const int gc   = wave & 7;            // col tile inside owned slice
  const int lane = threadIdx.x & 31;
  const int half = lane >> 4;
  const int lcol = lane & 15;
  const int colg = wg * 128 + gc * 16;  // global h-column tile base

  // zero-init state
  for (int i = threadIdx.x; i < B_ * H_; i += 512) h_lds[i] = (_Float16)0.0f;
  for (int i = threadIdx.x; i < B_ * 128; i += 512) c_lds[i] = 0.0f;
  __syncthreads();

  int par = 0;
#pragma unroll 1
  for (int s = 0; s < T_; ++s) {
    const int t = reverse ? (T_ - 1 - s) : s;

    // ---- recurrent GEMM: g = h @ Whh^T (owned cols, all 4 gates) ----------
    f8 acc[4];
#pragma unroll
    for (int gate = 0; gate < 4; ++gate) {
      int nt = (gate * H_ + colg) >> 4;
      __builtin_prefetch(
          (const void*)(whh_pk + (((size_t)0 * (G4H >> 4) + nt) * 32) * 16),
          0, 1);
      f8 a = {};
#pragma unroll 4
      for (int kb = 0; kb < (H_ >> 5); ++kb) {
        h16 af = load_a_tile((const _Float16*)h_lds, H_, gm * 16, kb * 32);
        h16 bf = load_b_packed(whh_pk, G4H >> 4, kb, nt);
        a = wmma16(af, bf, a);
      }
      acc[gate] = a;
    }

    // ---- gates + state update --------------------------------------------
    const int hcol = colg + lcol;       // owned global h column
    const int lc   = gc * 16 + lcol;    // local col in c slice
#pragma unroll
    for (int r = 0; r < 8; ++r) {
      int bidx = gm * 16 + r + 8 * half;              // D: M = r + 8*half
      size_t xb = ((size_t)t * B_ + bidx) * G4H;
      float iv = sigmoidf_(acc[0][r] + xg[xb + 0 * H_ + hcol]);
      float fv = sigmoidf_(acc[1][r] + xg[xb + 1 * H_ + hcol]);
      float gv = tanhf_   (acc[2][r] + xg[xb + 2 * H_ + hcol]);
      float ov = sigmoidf_(acc[3][r] + xg[xb + 3 * H_ + hcol]);
      float cv = fv * c_lds[bidx * 128 + lc] + iv * gv;
      c_lds[bidx * 128 + lc] = cv;
      float hv = ov * tanhf_(cv);
      _Float16 hh = (_Float16)hv;
      hstage[(size_t)par * B_ * H_ + (size_t)bidx * H_ + hcol] = hh;
      hout[((size_t)t * B_ + bidx) * H2 + dir * H_ + hcol] = hh;
    }

    // ---- cross-WGP exchange under cluster barrier ------------------------
    __threadfence();        // make hstage stores visible device-wide
    __syncthreads();
    __builtin_amdgcn_s_cluster_barrier();   // s_barrier_signal/-wait -3

    // TDM: pull the full (B x H) f16 h matrix from staging into LDS
    if (wave == 0) {
      tdm_load_to_lds(hstage + (size_t)par * B_ * H_, (void*)h_lds,
                      B_ * H_ * (unsigned)sizeof(_Float16));
      __builtin_amdgcn_s_wait_tensorcnt(0);
    }
    __syncthreads();
    par ^= 1;
  }
}

// ---------------------------------------------------------------------------
// 5) Classifier: logits(t,b,l) = h1 . cls_w[l] + cls_b[l]   (N=6: VALU dots)
// ---------------------------------------------------------------------------
__global__ void classifier(const _Float16* __restrict__ h1,   // (T*B, 1024)
                           const float* __restrict__ w,       // (6, 1024)
                           const float* __restrict__ bias,
                           float* __restrict__ logits) {      // (T*B, 6)
  int m = blockIdx.x * blockDim.x + threadIdx.x;
  if (m >= MROW) return;
  float acc[L_];
#pragma unroll
  for (int l = 0; l < L_; ++l) acc[l] = bias[l];
  const _Float16* hr = h1 + (size_t)m * H2;
  for (int k = 0; k < H2; ++k) {
    float hv = (float)hr[k];
#pragma unroll
    for (int l = 0; l < L_; ++l) acc[l] += hv * w[l * H2 + k];
  }
#pragma unroll
  for (int l = 0; l < L_; ++l) logits[(size_t)m * L_ + l] = acc[l];
}

// ---------------------------------------------------------------------------
// 6) CRF loss: 1 block, 32 threads (one batch element each), scalar out.
// ---------------------------------------------------------------------------
__global__ void crf_loss(const float* __restrict__ logits,    // (T*B, 6)
                         const int* __restrict__ amask,       // (B, T)
                         const int* __restrict__ labels,      // (B, T)
                         const float* __restrict__ cstart,
                         const float* __restrict__ cend,
                         const float* __restrict__ ctrans,    // (6,6)
                         float* __restrict__ out) {
  __shared__ float sm[B_];
  int b = threadIdx.x;
  float tr[L_ * L_];
#pragma unroll
  for (int i = 0; i < L_ * L_; ++i) tr[i] = ctrans[i];

  int lab0 = labels[b * T_ + 0];
  int tg0  = (lab0 == -100) ? 0 : lab0;
  float score = cstart[tg0] + logits[(size_t)b * L_ + tg0];
  int prev = tg0;

  float alpha[L_];
#pragma unroll
  for (int j = 0; j < L_; ++j) alpha[j] = cstart[j] + logits[(size_t)b * L_ + j];

#pragma unroll 1
  for (int t = 1; t < T_; ++t) {
    int lab  = labels[b * T_ + t];
    bool msk = (amask[b * T_ + t] != 0) && (lab != -100);
    int  tgt = (lab == -100) ? 0 : lab;
    const float* e = logits + ((size_t)t * B_ + b) * L_;
    if (msk) {
      score += tr[prev * L_ + tgt] + e[tgt];
      prev = tgt;
      float nxt[L_];
#pragma unroll
      for (int j = 0; j < L_; ++j) {
        float mx = -3.0e38f;
#pragma unroll
        for (int i = 0; i < L_; ++i)
          mx = fmaxf(mx, alpha[i] + tr[i * L_ + j]);
        float s = 0.0f;
#pragma unroll
        for (int i = 0; i < L_; ++i)
          s += __expf(alpha[i] + tr[i * L_ + j] - mx);
        nxt[j] = mx + __logf(s) + e[j];
      }
#pragma unroll
      for (int j = 0; j < L_; ++j) alpha[j] = nxt[j];
    }
  }
  score += cend[prev];
  float mx = -3.0e38f;
#pragma unroll
  for (int j = 0; j < L_; ++j) mx = fmaxf(mx, alpha[j] + cend[j]);
  float s = 0.0f;
#pragma unroll
  for (int j = 0; j < L_; ++j) s += __expf(alpha[j] + cend[j] - mx);
  float den = mx + __logf(s);

  sm[b] = score - den;
  __syncthreads();
  if (b == 0) {
    float acc = 0.0f;
#pragma unroll
    for (int i = 0; i < B_; ++i) acc += sm[i];
    out[0] = -acc / (float)B_;
  }
}

// ---------------------------------------------------------------------------
// Host orchestration
// ---------------------------------------------------------------------------
extern "C" void kernel_launch(void* const* d_in, const int* in_sizes, int n_in,
                              void* d_out, int out_size, void* d_ws, size_t ws_size,
                              hipStream_t stream) {
  const float* emb    = (const float*)d_in[0];
  const int*   amask  = (const int*)d_in[1];
  const int*   labels = (const int*)d_in[2];
  const float* c5w = (const float*)d_in[3];
  const float* c5b = (const float*)d_in[4];
  const float* c7w = (const float*)d_in[5];
  const float* c7b = (const float*)d_in[6];
  const float* c9w = (const float*)d_in[7];
  const float* c9b = (const float*)d_in[8];
  const float* bng = (const float*)d_in[9];
  const float* bnb = (const float*)d_in[10];
  const float* bnm = (const float*)d_in[11];
  const float* bnv = (const float*)d_in[12];
  const float* clw = (const float*)d_in[13];
  const float* clb = (const float*)d_in[14];
  const float* cst = (const float*)d_in[15];
  const float* cen = (const float*)d_in[16];
  const float* ctr = (const float*)d_in[17];
  struct WSet { const float *wih, *whh, *bih, *bhh; };
  WSet lw[4];
  for (int i = 0; i < 4; ++i) {
    lw[i].wih = (const float*)d_in[18 + 4 * i + 0];
    lw[i].whh = (const float*)d_in[18 + 4 * i + 1];
    lw[i].bih = (const float*)d_in[18 + 4 * i + 2];
    lw[i].bhh = (const float*)d_in[18 + 4 * i + 3];
  }
  float* out = (float*)d_out;

  // workspace carve-up
  char* p = (char*)d_ws;
  auto alloc = [&](size_t bytes) {
    void* r = (void*)p;
    p += (bytes + 255) & ~(size_t)255;
    return r;
  };
  _Float16* emb16  = (_Float16*)alloc((size_t)B_ * TP_ * E_ * 2);  // 34.6 MB
  _Float16* wc5    = (_Float16*)alloc((size_t)5 * E_ * C_ * 2);
  _Float16* wc7    = (_Float16*)alloc((size_t)9 * E_ * C_ * 2);
  _Float16* wc9    = (_Float16*)alloc((size_t)9 * E_ * C_ * 2);
  _Float16* xc16   = (_Float16*)alloc((size_t)MROW * D3C * 2);     // 25 MB
  _Float16* wih_pk = (_Float16*)alloc((size_t)G4H * 1024 * 2);     // 4.2 MB
  _Float16* whh_pk = (_Float16*)alloc((size_t)G4H * H_ * 2);       // 2 MB
  float*    xg     = (float*)   alloc((size_t)MROW * G4H * 4);     // 134 MB
  _Float16* hcat0  = (_Float16*)alloc((size_t)MROW * H2 * 2);      // 33.5 MB
  _Float16* hcat1  = (_Float16*)alloc((size_t)MROW * H2 * 2);      // 33.5 MB
  _Float16* hstage = (_Float16*)alloc((size_t)2 * B_ * H_ * 2);
  float*    logits = (float*)   alloc((size_t)MROW * L_ * 4);

  // 0) cast + zero-pad embeddings to f16 (B, TP_, E)
  {
    long n = (long)B_ * TP_ * E_;
    cast_pad_f16<<<(unsigned)((n + 255) / 256), 256, 0, stream>>>(emb, emb16);
  }

  // 1) pack conv weights per-tap into B-fragment layout
  {
    long n = (long)E_ * C_;
    unsigned g = (unsigned)((n + 255) / 256);
    for (int tap = 0; tap < 5; ++tap)
      pack_b_f16<<<g, 256, 0, stream>>>(c5w, wc5 + (size_t)tap * E_ * C_,
                                        E_, C_, 5, (long)E_ * 5, tap);
    for (int tap = 0; tap < 9; ++tap)
      pack_b_f16<<<g, 256, 0, stream>>>(c7w, wc7 + (size_t)tap * E_ * C_,
                                        E_, C_, 9, (long)E_ * 9, tap);
    for (int tap = 0; tap < 9; ++tap)
      pack_b_f16<<<g, 256, 0, stream>>>(c9w, wc9 + (size_t)tap * E_ * C_,
                                        E_, C_, 9, (long)E_ * 9, tap);
  }

  // 2) conv bank + BN + GELU -> xc16 (T*B, 768) f16
  conv_bn_gelu<<<dim3(T_ / 16, B_, D3C / 16), 32, 0, stream>>>(
      emb16, wc5, wc7, wc9, c5b, c7b, c9b, bng, bnb, bnm, bnv, xc16);

  // 3) two BiLSTM layers
  for (int li = 0; li < 2; ++li) {
    const _Float16* xin = (li == 0) ? xc16 : hcat0;
    int din             = (li == 0) ? D3C : H2;
    _Float16* hc        = (li == 0) ? hcat0 : hcat1;
    for (int dir = 0; dir < 2; ++dir) {
      const WSet& w = lw[li * 2 + dir];
      long n1 = (long)din * G4H;
      pack_b_f16<<<(unsigned)((n1 + 255) / 256), 256, 0, stream>>>(
          w.wih, wih_pk, din, G4H, 1, (long)din, 0);
      long n2 = (long)H_ * G4H;
      pack_b_f16<<<(unsigned)((n2 + 255) / 256), 256, 0, stream>>>(
          w.whh, whh_pk, H_, G4H, 1, (long)H_, 0);
      // xg = x @ Wih^T + (bih + bhh)   : M=16384, K=din, N=2048
      gemm_bias<<<dim3(MROW / 16, G4H / 128), 256, 0, stream>>>(
          xin, wih_pk, w.bih, w.bhh, xg, MROW, din, G4H);
      // sequential recurrence across a 4-WGP cluster
      lstm_scan<<<4, 512, 0, stream>>>(xg, whh_pk, hstage, hc, dir, dir);
    }
  }

  // 4) classifier logits (T*B, 6)
  classifier<<<(MROW + 255) / 256, 256, 0, stream>>>(hcat1, clw, clb, logits);

  // 5) CRF negative log-likelihood -> scalar
  crf_loss<<<1, B_, 0, stream>>>(logits, amask, labels, cst, cen, ctr, out);
}
